// CAM_21199958573260
// MI455X (gfx1250) — matrix operand order repaired and checked by reading, not compiled, not run on hardware
//
#include <hip/hip_runtime.h>
#include <hip/hip_bf16.h>
#include <math.h>

// ---------------- problem constants (from reference) ----------------
#define B_   2
#define N1_  75
#define N2_  25
#define C_   640
#define HW_  36
#define MP_  48      // HW padded up to 3 WMMA tiles of 16 (rows 36..47 zero)
#define KSTEPS_ (C_ / 32)   // 20 WMMA K-steps of 32

typedef __attribute__((ext_vector_type(16))) _Float16 v16h;
typedef __attribute__((ext_vector_type(8)))  _Float16 v8h;
typedef __attribute__((ext_vector_type(8)))  float    v8f;

union Frag16 { v16h v; v8h h[2]; };

// --------------------------------------------------------------------
// Kernel 1: L2-normalize over C (eps 1e-12, per reference) and transpose
// each (b,idx) slab from [C, HW] f32 to row-major [MP, C] f16 with zero
// padding rows 36..47.  This layout serves as WMMA A (f1) and B^T (f2).
// --------------------------------------------------------------------
__global__ __launch_bounds__(256) void norm_tr_kernel(
    const float* __restrict__ src, _Float16* __restrict__ dst) {
  const int blk = blockIdx.x;                 // b*n + idx (flattened)
  const float* s = src + (size_t)blk * C_ * HW_;
  _Float16*    d = dst + (size_t)blk * MP_ * C_;

  __shared__ float part[256];
  __shared__ float rnorm[HW_];
  const int t = threadIdx.x;

  // partial column sums of squares: 7 partials per column m
  float acc = 0.0f;
  if (t < 7 * HW_) {                          // 252 active
    const int m  = t % HW_;
    const int cg = t / HW_;                   // 0..6
    for (int c = cg; c < C_; c += 7) {
      const float v = s[c * HW_ + m];
      acc += v * v;
    }
  }
  part[t] = acc;
  __syncthreads();
  if (t < HW_) {
    float sum = 0.0f;
    #pragma unroll
    for (int k = 0; k < 7; ++k) sum += part[t + k * HW_];
    rnorm[t] = 1.0f / fmaxf(sqrtf(sum), 1e-12f);
  }
  __syncthreads();

  // transposed, normalized, f16, zero-padded write: d[m*C_ + c]
  for (int o = t; o < MP_ * C_; o += 256) {
    const int m = o / C_;
    const int c = o - m * C_;
    const float v = (m < HW_) ? s[c * HW_ + m] * rnorm[m] : 0.0f;
    d[o] = (_Float16)v;
  }
}

// --------------------------------------------------------------------
// Kernel 2: per (b,i,j) workgroup (3 waves = 96 threads):
//   - 48x48 (valid 36x36) GEMM  a = f1n[b,i]^T x f2n[b,j]  via
//     v_wmma_f32_16x16x32_f16, K = 640
//   - full dual attention (a and a^T) + softmax blend -> Aout[b,i,j,36]
// --------------------------------------------------------------------
__global__ __launch_bounds__(96) void attn_kernel(
    const _Float16* __restrict__ f1h, const _Float16* __restrict__ f2h,
    const float* __restrict__ alpha_p,
    const float* __restrict__ W1, const float* __restrict__ b1,
    const float* __restrict__ W2, const float* __restrict__ b2,
    float* __restrict__ Aout) {
  const int blk = blockIdx.x;                      // ((b*N1+i)*N2+j)
  const int j = blk % N2_;
  const int i = (blk / N2_) % N1_;
  const int b = blk / (N1_ * N2_);

  const int t = threadIdx.x;
  const int w    = t >> 5;                         // wave id 0..2 = M-tile
  const int L    = t & 31;                         // lane
  const int lhalf = L >> 4;                        // lane group 0/1
  const int lmod  = L & 15;                        // row/col within tile

  __shared__ float aLds[HW_ * 37];                 // pitch 37: conflict-free-ish
  __shared__ float am1[HW_], am2[HW_];
  __shared__ float g1[48], g2[48];
  __shared__ float h1[HW_], h2[HW_];
  __shared__ float s1v[HW_], s2v[HW_];

  // Per-lane row pointers (padded rows are zero, so no edge masking needed;
  // EXEC stays all-ones through the WMMA loop).
  const _Float16* Arow = f1h + ((size_t)(b * N1_ + i) * MP_ + (16 * w + lmod)) * C_;
  const _Float16* Brow = f2h + ((size_t)(b * N2_ + j) * MP_ + lmod) * C_;

  v8f c0 = {}, c1 = {}, c2 = {};
  for (int ks = 0; ks < KSTEPS_; ++ks) {
    const int kb = ks * 32;
    Frag16 a, q0, q1, q2;
    // A 16x32 fragment: ISA layout -> lane g=0: K {0..7}U{16..23}; g=1: +8.
    // Two b128 loads at K offsets (kb + 8*g) and (kb + 16 + 8*g).
    const _Float16* ap = Arow + kb + 8 * lhalf;
    a.h[0] = *(const v8h*)(ap);
    a.h[1] = *(const v8h*)(ap + 16);
    // B 32x16 fragment: same K layout as A -> lane g holds K {8g..8g+7}U{16+8g..16+8g+7}
    // of col n, i.e. two b128 loads of B^T row n at kb + 8*g and kb + 16 + 8*g.
    const _Float16* bp = Brow + kb + 8 * lhalf;
    q0.h[0] = *(const v8h*)(bp);
    q0.h[1] = *(const v8h*)(bp + 16);
    q1.h[0] = *(const v8h*)(bp + 16 * C_);
    q1.h[1] = *(const v8h*)(bp + 16 * C_ + 16);
    q2.h[0] = *(const v8h*)(bp + 32 * C_);
    q2.h[1] = *(const v8h*)(bp + 32 * C_ + 16);

    c0 = __builtin_amdgcn_wmma_f32_16x16x32_f16(false, a.v, false, q0.v,
                                                (short)0, c0, false, false);
    c1 = __builtin_amdgcn_wmma_f32_16x16x32_f16(false, a.v, false, q1.v,
                                                (short)0, c1, false, false);
    c2 = __builtin_amdgcn_wmma_f32_16x16x32_f16(false, a.v, false, q2.v,
                                                (short)0, c2, false, false);
  }

  // Scatter C tiles into LDS (C layout: VGPR r -> row r + 8*laneGroup, col lmod)
  #pragma unroll
  for (int r = 0; r < 8; ++r) {
    const int m = 16 * w + 8 * lhalf + r;
    if (m < HW_) {
      aLds[m * 37 + lmod]      = c0[r];
      aLds[m * 37 + 16 + lmod] = c1[r];
      if (32 + lmod < HW_) aLds[m * 37 + 32 + lmod] = c2[r];
    }
  }
  __syncthreads();

  // ---- attention on the 36x36 matrix, both orientations at once ----
  // am1[n] = col-mean of a (a1 path);  am2[n] = row-mean of a (a2 path,
  // and also the per-row mu for a1's normalization).
  if (t < HW_) {
    float cs = 0.0f, rs = 0.0f;
    for (int k = 0; k < HW_; ++k) {
      cs += aLds[k * 37 + t];
      rs += aLds[t * 37 + k];
    }
    am1[t] = cs * (1.0f / HW_);
    am2[t] = rs * (1.0f / HW_);
  }
  __syncthreads();

  if (t < 48) {                                   // g = relu(W1 @ am + b1)
    float a1acc = b1[t], a2acc = b1[t];
    for (int h = 0; h < HW_; ++h) {
      const float wv = W1[t * HW_ + h];
      a1acc += wv * am1[h];
      a2acc += wv * am2[h];
    }
    g1[t] = fmaxf(a1acc, 0.0f);
    g2[t] = fmaxf(a2acc, 0.0f);
  }
  __syncthreads();

  if (t < HW_) {                                  // h = W2 @ g + b2
    float a1acc = b2[t], a2acc = b2[t];
    for (int o = 0; o < 48; ++o) {
      const float wv = W2[t * 48 + o];
      a1acc += wv * g1[o];
      a2acc += wv * g2[o];
    }
    h1[t] = a1acc;
    h2[t] = a2acc;
  }
  __syncthreads();

  if (t < HW_) {
    // a1 path: row t of a, mu = row mean, std with ddof=1
    float mu = am2[t], var = 0.0f, s = 0.0f;
    for (int n = 0; n < HW_; ++n) {
      const float d = aLds[t * 37 + n] - mu;
      var += d * d;
    }
    const float sd = sqrtf(var * (1.0f / (HW_ - 1))) + 1e-6f;
    for (int n = 0; n < HW_; ++n) s += (aLds[t * 37 + n] - mu) * h1[n];
    s1v[t] = s / (HW_ * sd);

    // a2 path: column t of a
    float mu2 = am1[t], var2 = 0.0f, s2 = 0.0f;
    for (int n = 0; n < HW_; ++n) {
      const float d = aLds[n * 37 + t] - mu2;
      var2 += d * d;
    }
    const float sd2 = sqrtf(var2 * (1.0f / (HW_ - 1))) + 1e-6f;
    for (int n = 0; n < HW_; ++n) s2 += (aLds[n * 37 + t] - mu2) * h2[n];
    s2v[t] = s2 / (HW_ * sd2);
  }
  __syncthreads();

  if (t < HW_) {                                  // softmax(s/0.025)+1, blend
    float mx1 = -1e30f, mx2 = -1e30f;
    for (int k = 0; k < HW_; ++k) {
      mx1 = fmaxf(mx1, s1v[k]);
      mx2 = fmaxf(mx2, s2v[k]);
    }
    float z1 = 0.0f, z2 = 0.0f;
    for (int k = 0; k < HW_; ++k) {
      z1 += __expf((s1v[k] - mx1) * 40.0f);
      z2 += __expf((s2v[k] - mx2) * 40.0f);
    }
    const float p1 = __expf((s1v[t] - mx1) * 40.0f) / z1 + 1.0f;
    const float p2 = __expf((s2v[t] - mx2) * 40.0f) / z2 + 1.0f;
    const float al = 1.0f / (1.0f + __expf(-alpha_p[0]));
    Aout[(size_t)blk * HW_ + t] = al * p1 + (1.0f - al) * p2;
  }
}

// --------------------------------------------------------------------
// Kernel 3: HBM-bound expansion. One block per (b,i,j): stream
//   f1o[b,j,i,c,m] = f1[b,i,c,m] * (1 + A[b,i,j,m])
//   f2o[b,j,i,c,m] = f2[b,j,c,m] * (1 + A[b,i,j,m])
// Fully coalesced reads and writes; A broadcast via LDS.
// --------------------------------------------------------------------
__global__ __launch_bounds__(256) void expand_kernel(
    const float* __restrict__ f1, const float* __restrict__ f2,
    const float* __restrict__ Aout,
    float* __restrict__ out1, float* __restrict__ out2) {
  const int blk = blockIdx.x;                      // ((b*N1+i)*N2+j)
  const int j = blk % N2_;
  const int i = (blk / N2_) % N1_;
  const int b = blk / (N1_ * N2_);

  __shared__ float Am[HW_];
  const int t = threadIdx.x;
  if (t < HW_) Am[t] = 1.0f + Aout[(size_t)blk * HW_ + t];
  __syncthreads();

  const float* s1 = f1 + (size_t)(b * N1_ + i) * C_ * HW_;
  const float* s2 = f2 + (size_t)(b * N2_ + j) * C_ * HW_;
  const size_t dbase = (size_t)((b * N2_ + j) * N1_ + i) * C_ * HW_;
  float* d1 = out1 + dbase;
  float* d2 = out2 + dbase;

  for (int o = t; o < C_ * HW_; o += 256) {
    const float a = Am[o % HW_];
    d1[o] = s1[o] * a;
    d2[o] = s2[o] * a;
  }
}

// --------------------------------------------------------------------
extern "C" void kernel_launch(void* const* d_in, const int* in_sizes, int n_in,
                              void* d_out, int out_size, void* d_ws, size_t ws_size,
                              hipStream_t stream) {
  const float* f1    = (const float*)d_in[0];   // [2,75,640,36]
  const float* f2    = (const float*)d_in[1];   // [2,25,640,36]
  const float* alpha = (const float*)d_in[2];   // scalar
  const float* W1    = (const float*)d_in[3];   // [48,36]
  const float* b1    = (const float*)d_in[4];   // [48]
  const float* W2    = (const float*)d_in[5];   // [36,48]
  const float* b2    = (const float*)d_in[6];   // [36]

  // workspace layout (256B aligned):
  //   f1h : [2,75,48,640] f16 = 9,216,000 B
  //   f2h : [2,25,48,640] f16 = 3,072,000 B
  //   A   : [2,75,25,36]  f32 =   540,000 B
  char* ws = (char*)d_ws;
  _Float16* f1h  = (_Float16*)(ws);
  _Float16* f2h  = (_Float16*)(ws + 9216000);
  float*    Aatt = (float*)   (ws + 12288000);

  float* out1 = (float*)d_out;                        // f1o: 86,400,000 floats
  float* out2 = (float*)d_out + (size_t)B_ * N2_ * N1_ * C_ * HW_;

  norm_tr_kernel<<<B_ * N1_, 256, 0, stream>>>(f1, f1h);
  norm_tr_kernel<<<B_ * N2_, 256, 0, stream>>>(f2, f2h);
  attn_kernel<<<B_ * N1_ * N2_, 96, 0, stream>>>(f1h, f2h, alpha,
                                                 W1, b1, W2, b2, Aatt);
  expand_kernel<<<B_ * N1_ * N2_, 256, 0, stream>>>(f1, f2, Aatt, out1, out2);
}
